// QuantizedClassifier_52381421142731
// MI455X (gfx1250) — matrix operand order repaired and checked by reading, not compiled, not run on hardware
//
#include <hip/hip_runtime.h>
#include <hip/hip_bf16.h>

typedef _Float16 v16h __attribute__((ext_vector_type(16)));
typedef float    v8f  __attribute__((ext_vector_type(8)));
typedef _Float16 h8   __attribute__((ext_vector_type(8)));
typedef _Float16 h4   __attribute__((ext_vector_type(4)));
typedef int      i4v  __attribute__((ext_vector_type(4)));

#define TILE_M 64
#define TILE_N 128
#define TILE_K 64
#define NCENTERS 64

// --- async copy (global -> LDS), probe-verified on this toolchain -----------
#if __has_builtin(__builtin_amdgcn_global_load_async_to_lds_b128)
#define HAVE_ASYNC_LDS 1
typedef __attribute__((address_space(3))) i4v* lds_i4p;
static __device__ __forceinline__ void async_cp_b128(const void* gsrc, void* ldst) {
    __builtin_amdgcn_global_load_async_to_lds_b128(
        (i4v*)gsrc, (lds_i4p)ldst, 0, 0);
}
static __device__ __forceinline__ void wait_async_lds() {
#if __has_builtin(__builtin_amdgcn_s_wait_asynccnt)
    __builtin_amdgcn_s_wait_asynccnt(0);
#else
    asm volatile("s_wait_asynccnt 0" ::: "memory");
#endif
}
#else
#define HAVE_ASYNC_LDS 0
#endif

// ---------------------------------------------------------------------------
// Kernel 1: scalar quantization. Nearest of 64 centers (first-min tie-break,
// matching jnp.argmin); emit f16.
// ---------------------------------------------------------------------------
__global__ void __launch_bounds__(256)
quantize_kernel(const float* __restrict__ x, const float* __restrict__ centers,
                _Float16* __restrict__ qA, int n4)
{
    __shared__ float sc[NCENTERS];
    if (threadIdx.x < NCENTERS) sc[threadIdx.x] = centers[threadIdx.x];
    __syncthreads();

    int idx = blockIdx.x * blockDim.x + threadIdx.x;
    if (idx >= n4) return;

    float4 xv = ((const float4*)x)[idx];
    float vals[4] = {xv.x, xv.y, xv.z, xv.w};
    h4 q;
#pragma unroll
    for (int j = 0; j < 4; ++j) {
        float v    = vals[j];
        float best = sc[0];
        float bd   = (v - sc[0]) * (v - sc[0]);
#pragma unroll 8
        for (int c = 1; c < NCENTERS; ++c) {
            float d = (v - sc[c]) * (v - sc[c]);
            if (d < bd) { bd = d; best = sc[c]; }
        }
        q[j] = (_Float16)best;
    }
    ((h4*)qA)[idx] = q;
}

// ---------------------------------------------------------------------------
// Kernel 2: transpose + convert W (K x Nout, f32) -> Wt (Npad x K, f16).
// Tiled through LDS so global reads and writes both coalesce. One-time prep.
// ---------------------------------------------------------------------------
#define TT_K 32
#define TT_N 64
__global__ void __launch_bounds__(256)
transpose_w(const float* __restrict__ W, _Float16* __restrict__ Wt,
            int K, int Nout, int Npad)
{
    __shared__ float s[TT_K][TT_N + 4];
    const int k0 = blockIdx.x * TT_K;
    const int n0 = blockIdx.y * TT_N;

#pragma unroll
    for (int r = 0; r < 2; ++r) {
        int c  = threadIdx.x + r * 256;
        int k  = c >> 4;
        int nn = (c & 15) << 2;
        int gn = n0 + nn;
        float4 v;
        if (gn + 4 <= Nout) {
            v = *(const float4*)(W + (size_t)(k0 + k) * Nout + gn);
        } else {
            float t0 = (gn + 0 < Nout) ? W[(size_t)(k0 + k) * Nout + gn + 0] : 0.f;
            float t1 = (gn + 1 < Nout) ? W[(size_t)(k0 + k) * Nout + gn + 1] : 0.f;
            float t2 = (gn + 2 < Nout) ? W[(size_t)(k0 + k) * Nout + gn + 2] : 0.f;
            float t3 = (gn + 3 < Nout) ? W[(size_t)(k0 + k) * Nout + gn + 3] : 0.f;
            v = make_float4(t0, t1, t2, t3);
        }
        s[k][nn + 0] = v.x; s[k][nn + 1] = v.y;
        s[k][nn + 2] = v.z; s[k][nn + 3] = v.w;
    }
    __syncthreads();

    const int n  = threadIdx.x >> 2;
    const int ks = (threadIdx.x & 3) << 3;
    h8 o;
#pragma unroll
    for (int j = 0; j < 8; ++j) o[j] = (_Float16)s[ks + j][n];
    *(h8*)(Wt + (size_t)(n0 + n) * K + k0 + ks) = o;
}

// ---------------------------------------------------------------------------
// Kernel 3: f16 WMMA GEMM, f32 accumulation, fused bias.
//   out(M x Nout) = qA(M x K) * Wt(Npad x K)^T + bias
// Block = 256 threads (8 waves), tile 64x128, K-step 64, double-buffered LDS.
// Pipeline: wait tile i -> barrier -> issue async fill of tile i+1 -> compute
// tile i. Fragment loads are batched ahead of the 4 WMMAs of each K-half so
// the waitcnt pass can emit partial s_wait_dscnt and hide LDS latency.
// ---------------------------------------------------------------------------
__global__ void __launch_bounds__(256)
gemm_wmma(const _Float16* __restrict__ A, const _Float16* __restrict__ Bt,
          const float* __restrict__ bias, float* __restrict__ out,
          int M, int Npad, int K, int Nout)
{
    __shared__ _Float16 sA [2][TILE_M][TILE_K];   // 2 x 8 KB
    __shared__ _Float16 sBt[2][TILE_N][TILE_K];   // 2 x 16 KB

    const int tid  = threadIdx.x;
    const int lane = tid & 31;
    const int wave = tid >> 5;

    const int bm = blockIdx.x * TILE_M;
    const int bn = blockIdx.y * TILE_N;

    const int mOff   = (wave & 3) * 16;    // 4 M-subtiles
    const int nGroup = (wave >> 2) * 64;   // 2 N-groups of 64

    // fragment geometry (ISA 7.12.2, 16-bit layouts, wave32)
    const int laneM  = lane & 15;
    const int laneHi = lane >> 4;
    const int aKlo = laneHi * 8;
    const int aKhi = 16 + laneHi * 8;
    const int bK   = laneHi * 8;

    v8f acc[4] = {v8f{}, v8f{}, v8f{}, v8f{}};

    auto issue_tile = [&](int buf, int k0) {
        // A tile: 512 b128 chunks, 2 per thread
#pragma unroll
        for (int r = 0; r < 2; ++r) {
            int c   = tid + r * 256;
            int row = c >> 3, seg = (c & 7) << 3;
            const _Float16* gp = A + (size_t)(bm + row) * K + k0 + seg;
#if HAVE_ASYNC_LDS
            async_cp_b128(gp, &sA[buf][row][seg]);
#else
            *(h8*)&sA[buf][row][seg] = *(const h8*)gp;
#endif
        }
        // B tile (pre-transposed in global): 1024 chunks, 4 per thread
#pragma unroll
        for (int r = 0; r < 4; ++r) {
            int c   = tid + r * 256;
            int row = c >> 3, seg = (c & 7) << 3;
            const _Float16* gp = Bt + (size_t)(bn + row) * K + k0 + seg;
#if HAVE_ASYNC_LDS
            async_cp_b128(gp, &sBt[buf][row][seg]);
#else
            *(h8*)&sBt[buf][row][seg] = *(const h8*)gp;
#endif
        }
    };

    const int nt = K / TILE_K;
    issue_tile(0, 0);

    for (int i = 0; i < nt; ++i) {
        const int cur = i & 1;
#if HAVE_ASYNC_LDS
        wait_async_lds();        // my share of tile i has landed in LDS
#endif
        __syncthreads();         // tile i visible to all; tile i-1 fully consumed

        if (i + 1 < nt) issue_tile(cur ^ 1, (i + 1) * TILE_K);  // overlaps compute

        // ---- compute tile i: 2 K-halves x 4 N-subtiles = 8 WMMAs ----
#pragma unroll
        for (int kk = 0; kk < TILE_K; kk += 32) {
            // batch ALL fragment loads for this K-half first, so the compiler
            // can interleave partial s_wait_dscnt with the WMMA sequence
            h8 alo = *(const h8*)&sA[cur][mOff + laneM][kk + aKlo];
            h8 ahi = *(const h8*)&sA[cur][mOff + laneM][kk + aKhi];
            h8 blo[4], bhi[4];
#pragma unroll
            for (int t = 0; t < 4; ++t) {
                const int nRow = nGroup + t * 16 + laneM;
                blo[t] = *(const h8*)&sBt[cur][nRow][kk + bK];
                bhi[t] = *(const h8*)&sBt[cur][nRow][kk + bK + 16];
            }

            v16h afrag;
#pragma unroll
            for (int j = 0; j < 8; ++j) { afrag[j] = alo[j]; afrag[8 + j] = ahi[j]; }

#pragma unroll
            for (int t = 0; t < 4; ++t) {
                v16h bfrag;
#pragma unroll
                for (int j = 0; j < 8; ++j) { bfrag[j] = blo[t][j]; bfrag[8 + j] = bhi[t][j]; }
                acc[t] = __builtin_amdgcn_wmma_f32_16x16x32_f16(
                    false, afrag, false, bfrag, (short)0, acc[t], false, false);
            }
        }
    }

    // ---- store: C/D layout row = v + 8*laneHi, col = laneM; fused bias ----
#pragma unroll
    for (int t = 0; t < 4; ++t) {
        const int col = bn + nGroup + t * 16 + laneM;
        if (col < Nout) {
            const float bv = bias[col];
#pragma unroll
            for (int v = 0; v < 8; ++v) {
                int row = bm + mOff + 8 * laneHi + v;
                out[(size_t)row * Nout + col] = acc[t][v] + bv;
            }
        }
    }
}

// ---------------------------------------------------------------------------
// Host launcher
// ---------------------------------------------------------------------------
extern "C" void kernel_launch(void* const* d_in, const int* in_sizes, int n_in,
                              void* d_out, int out_size, void* d_ws, size_t ws_size,
                              hipStream_t stream)
{
    const float* x       = (const float*)d_in[0];   // (M, K) f32
    const float* centers = (const float*)d_in[1];   // (64, 1) f32
    const float* W       = (const float*)d_in[2];   // (K, Nout) f32
    const float* bias    = (const float*)d_in[3];   // (Nout,) f32
    float* out           = (float*)d_out;           // (M, Nout) f32

    const int Nout = in_sizes[3];                   // 1000
    const int K    = in_sizes[2] / Nout;            // 3072
    const int M    = in_sizes[0] / K;               // 512
    const int Npad = (Nout + 127) & ~127;           // 1024

    _Float16* qA = (_Float16*)d_ws;                 // M*K halfs
    _Float16* Wt = qA + (size_t)M * K;              // Npad*K halfs (transposed)

    int n4 = (M * K) / 4;
    quantize_kernel<<<(n4 + 255) / 256, 256, 0, stream>>>(x, centers, qA, n4);

    dim3 tgrid(K / TT_K, Npad / TT_N);
    transpose_w<<<tgrid, 256, 0, stream>>>(W, Wt, K, Nout, Npad);

    dim3 grid(M / TILE_M, Npad / TILE_N);
    gemm_wmma<<<grid, 256, 0, stream>>>(qA, Wt, bias, out, M, Npad, K, Nout);
}
